// Tail_18236431139553
// MI455X (gfx1250) — matrix operand order repaired and checked by reading, not compiled, not run on hardware
//
#include <hip/hip_runtime.h>
#include <math.h>

typedef __bf16 bf16;
typedef __attribute__((ext_vector_type(16))) __bf16 v16bf;
typedef __attribute__((ext_vector_type(8)))  float  v8f;
typedef __attribute__((ext_vector_type(4)))  float  f32x4;

union FragBF { v16bf v; uint4 u[2]; };

// ---------------- constants ----------------
#define KTOT    100352            // 2048*49
#define KSPLIT  32
#define KCHUNK  (KTOT / KSPLIT)   // 3136 = 98*32
#define BKQ     32                // K-tile per LDS stage
#define NITER   (KCHUNK / BKQ)    // 98
#define TILE_M  64
#define TILE_N  256
#define LDK     40                // padded LDS row stride (elements) = 80B: 16B-aligned, conflict-free

// ---------------- kernel 1: PE/beta bias table [32][2048] ----------------
__global__ __launch_bounds__(256) void prep_bias(const float* __restrict__ beta,
                                                 float* __restrict__ biastab) {
  int g = blockIdx.x * 256 + threadIdx.x;       // t*2048 + c, 65536 total
  int t = g >> 11, c = g & 2047;
  float freq = exp2f(-(float)c * (13.2877123795494f / 1024.0f)); // log2(10000)/1024
  float arg  = (float)t * freq;
  float pe   = (c & 1) ? cosf(arg) : sinf(arg);
  biastab[g] = beta[c] * 45.254833995939045f + pe;   // sqrt(2048)
}

// ---------------- kernel 2: xs[n,k] = x*A[c] + bias[t,c]  (bf16) ----------------
__global__ __launch_bounds__(256) void prep_x(const float* __restrict__ x,
                                              const float* __restrict__ gamma,
                                              const float* __restrict__ biastab,
                                              bf16* __restrict__ xs) {
  int kidx = blockIdx.x * 256 + threadIdx.x;    // 0..100351 (392 blocks)
  int n    = blockIdx.y;                        // 0..255
  int c    = kidx / 49;
  int t    = n & 31;
  float v = x[(size_t)n * KTOT + kidx] * (gamma[c] * 45.25460772f) + biastab[t * 2048 + c];
  xs[(size_t)n * KTOT + kidx] = (bf16)v;
}

// ---------------- kernel 3: split-K bf16 WMMA GEMM (double-buffered, async B) ----------------
// C_partial[bk][m][n] = sum_{k in chunk} W[m,k] * xs[n,k]
__global__ __launch_bounds__(256) void gemm_splitk(const float* __restrict__ W,
                                                   const bf16* __restrict__ X,
                                                   float* __restrict__ part) {
  __shared__ __align__(16) bf16 As[2][TILE_M * LDK];   //  64 x 32 (+pad) x2  = 10 KB
  __shared__ __align__(16) bf16 Bs[2][TILE_N * LDK];   // 256 x 32 (+pad) x2  = 40 KB

  const int tid = threadIdx.x;
  const int bm  = blockIdx.x;         // 0..15
  const int bk  = blockIdx.y;         // 0..31
  const int k0base = bk * KCHUNK;

  const int wave  = tid >> 5;
  const int lane  = tid & 31;
  const int wm    = wave & 1;         // M sub-tile: wm*32
  const int wn    = wave >> 1;        // N sub-tile: wn*64
  const int l16   = lane & 15;
  const int lhalf = lane >> 4;        // 0/1

  v8f acc[2][4];
  #pragma unroll
  for (int a = 0; a < 2; a++)
    #pragma unroll
    for (int b = 0; b < 4; b++)
      #pragma unroll
      for (int e = 0; e < 8; e++) acc[a][b][e] = 0.0f;

  f32x4 a_pre[2];

  // ---- B tile: 256x32 bf16 = 16KB, 4 async b128 per thread, direct global->LDS ----
  auto issueB = [&](int buf, int k0) {
    #pragma unroll
    for (int j = 0; j < 4; j++) {
      int idx = tid + j * 256;                   // 0..1023 16B chunks
      int row = idx >> 2;                        // 0..255
      int c8  = (idx & 3) << 3;                  // 0,8,16,24
      unsigned lds  = (unsigned)(uintptr_t)&Bs[buf][row * LDK + c8];
      unsigned goff = (unsigned)(((size_t)row * KTOT + k0 + c8) * 2u);
      asm volatile("global_load_async_to_lds_b128 %0, %1, %2"
                   :: "v"(lds), "v"(goff), "s"(X) : "memory");
    }
  };
  // ---- A tile: 64x32 fp32 = 8KB, prefetch to regs (non-temporal: W streamed once) ----
  auto loadA = [&](int k0) {
    #pragma unroll
    for (int j = 0; j < 2; j++) {
      int idx = tid + j * 256;                   // 0..511 float4 slots
      int row = idx >> 3;                        // 0..63
      int c4  = (idx & 7) << 2;                  // 0..28
      a_pre[j] = __builtin_nontemporal_load(
          (const f32x4*)(W + (size_t)(bm * TILE_M + row) * KTOT + k0 + c4));
    }
  };
  auto storeA = [&](int buf) {
    #pragma unroll
    for (int j = 0; j < 2; j++) {
      int idx = tid + j * 256;
      int row = idx >> 3;
      int c4  = (idx & 7) << 2;
      bf16* d = &As[buf][row * LDK + c4];
      d[0] = (bf16)a_pre[j].x; d[1] = (bf16)a_pre[j].y;
      d[2] = (bf16)a_pre[j].z; d[3] = (bf16)a_pre[j].w;
    }
  };
  auto compute = [&](int buf) {
    // A frag: lane=row(M); elems0-7 = K kb..kb+7, elems8-15 = K kb+16..kb+23, kb = lhalf*8
    const int kbA = lhalf * 8;
    FragBF af[2];
    #pragma unroll
    for (int a = 0; a < 2; a++) {
      const bf16* p = &As[buf][(wm * 32 + a * 16 + l16) * LDK + kbA];
      af[a].u[0] = *(const uint4*)(p);
      af[a].u[1] = *(const uint4*)(p + 16);
    }
    // B frag: lane=col(N); elems0-7 = K kb..kb+7, elems8-15 = K kb+16..kb+23, kb = lhalf*8
    const int kbB = lhalf * 8;
    #pragma unroll
    for (int b = 0; b < 4; b++) {
      const bf16* p = &Bs[buf][(wn * 64 + b * 16 + l16) * LDK + kbB];
      FragBF bfv;
      bfv.u[0] = *(const uint4*)(p);
      bfv.u[1] = *(const uint4*)(p + 16);
      #pragma unroll
      for (int a = 0; a < 2; a++)
        acc[a][b] = __builtin_amdgcn_wmma_f32_16x16x32_bf16(
            false, af[a].v, false, bfv.v, (short)0, acc[a][b], false, false);
    }
  };

  // prologue: fill buffer 0
  issueB(0, k0base);
  loadA(k0base);
  storeA(0);
  asm volatile("s_wait_asynccnt 0x0" ::: "memory");
  __syncthreads();

  for (int it = 0; it < NITER; ++it) {
    const int cur = it & 1;
    if (it + 1 < NITER) {
      issueB(cur ^ 1, k0base + (it + 1) * BKQ);   // async global->LDS, overlaps compute
      loadA(k0base + (it + 1) * BKQ);             // global->VGPR, overlaps compute
    }
    compute(cur);
    if (it + 1 < NITER) storeA(cur ^ 1);          // convert+store after WMMAs issued
    asm volatile("s_wait_asynccnt 0x0" ::: "memory");
    __syncthreads();
  }

  // ---- store partials: part[bk][M=1024][N=256]; D rows: M = r (+8 for hi lane half) ----
  #pragma unroll
  for (int a = 0; a < 2; a++)
    #pragma unroll
    for (int b = 0; b < 4; b++) {
      int n     = wn * 64 + b * 16 + l16;
      int mbase = bm * TILE_M + wm * 32 + a * 16 + lhalf * 8;
      float* o = part + ((size_t)bk * 1024 + mbase) * 256 + n;
      #pragma unroll
      for (int r = 0; r < 8; r++) o[(size_t)r * 256] = acc[a][b][r];
    }
}

// ---------------- kernel 4: reduce split-K + relu, transpose to xq[n][o] ----------------
__global__ __launch_bounds__(256) void reduce_relu(const float* __restrict__ part,
                                                   float* __restrict__ xq) {
  int g = blockIdx.x * 256 + threadIdx.x;   // 262144
  int n = g & 255, o = g >> 8;
  float s = 0.0f;
  #pragma unroll
  for (int ks = 0; ks < KSPLIT; ks++) s += part[((size_t)ks * 1024 + o) * 256 + n];
  xq[(size_t)n * 1024 + o] = s > 0.0f ? s : 0.0f;
}

// ---------------- kernel 5: layernorm over D=1024 (a=1, b=0, ddof=1) ----------------
__global__ __launch_bounds__(256) void layernorm_xq(const float* __restrict__ xq,
                                                    float* __restrict__ xqn) {
  __shared__ float red[256];
  int n = blockIdx.x, tid = threadIdx.x;
  const float* x = xq + (size_t)n * 1024;
  float vals[4], s = 0.0f;
  #pragma unroll
  for (int m = 0; m < 4; m++) { vals[m] = x[tid + m * 256]; s += vals[m]; }
  red[tid] = s; __syncthreads();
  for (int st = 128; st > 0; st >>= 1) { if (tid < st) red[tid] += red[tid + st]; __syncthreads(); }
  float mu = red[0] / 1024.0f; __syncthreads();
  float s2 = 0.0f;
  #pragma unroll
  for (int m = 0; m < 4; m++) { float v = vals[m] - mu; s2 += v * v; }
  red[tid] = s2; __syncthreads();
  for (int st = 128; st > 0; st >>= 1) { if (tid < st) red[tid] += red[tid + st]; __syncthreads(); }
  float inv = 1.0f / (sqrtf(red[0] / 1023.0f) + 1e-6f);
  float* o = xqn + (size_t)n * 1024;
  #pragma unroll
  for (int m = 0; m < 4; m++) o[tid + m * 256] = (vals[m] - mu) * inv;
}

// ---------------- kernel 6: 3-iteration attention tail, one block per batch ----------------
__global__ __launch_bounds__(256) void attn_tail(
    const float* __restrict__ xqn,
    const float* __restrict__ qlin_w, const float* __restrict__ qlin_b,
    const float* __restrict__ n1_a, const float* __restrict__ n1_b,
    const float* __restrict__ n2_a, const float* __restrict__ n2_b,
    const float* __restrict__ ff_w1, const float* __restrict__ ff_b1,
    const float* __restrict__ ff_w2, const float* __restrict__ ff_b2,
    float* __restrict__ out) {
  __shared__ float q[1024], qh[1024], z[1024], q_[1024];
  __shared__ float sc[16 * 32], hid[16 * 32], red[256];
  const int b = blockIdx.x, tid = threadIdx.x;
  const float* kbase = xqn + (size_t)b * 32 * 1024;     // kh = vh rows

  for (int j = tid; j < 1024; j += 256) q[j] = kbase[16 * 1024 + j];  // t//2 = 16
  __syncthreads();

  for (int i = 0; i < 3; i++) {
    // qh = relu(q @ W^T + bias)
    #pragma unroll
    for (int m = 0; m < 4; m++) {
      int o = tid + m * 256;
      const float* w = qlin_w + ((size_t)i * 1024 + o) * 1024;
      float s = 0.0f;
      for (int k = 0; k < 1024; k += 4) {
        float4 wv = *(const float4*)(w + k);
        s += wv.x * q[k] + wv.y * q[k + 1] + wv.z * q[k + 2] + wv.w * q[k + 3];
      }
      s += qlin_b[i * 1024 + o];
      qh[o] = s > 0.0f ? s : 0.0f;
    }
    __syncthreads();
    // scores[h][t] = (qh_h . k_{t,h}) / 8
    {
      int h = tid >> 4, tt = tid & 15;
      #pragma unroll
      for (int rep = 0; rep < 2; rep++) {
        int t = tt + rep * 16;
        const float* kv = kbase + (size_t)t * 1024 + h * 64;
        float s = 0.0f;
        for (int d = 0; d < 64; d++) s += qh[h * 64 + d] * kv[d];
        sc[h * 32 + t] = s * 0.125f;
      }
    }
    __syncthreads();
    // softmax per head
    if (tid < 16) {
      int h = tid; float mx = -1e30f;
      for (int t = 0; t < 32; t++) mx = fmaxf(mx, sc[h * 32 + t]);
      float sum = 0.0f;
      for (int t = 0; t < 32; t++) { float e = expf(sc[h * 32 + t] - mx); sc[h * 32 + t] = e; sum += e; }
      float inv = 1.0f / sum;
      for (int t = 0; t < 32; t++) sc[h * 32 + t] *= inv;
    }
    __syncthreads();
    // A = sc @ v ; z = A + qh
    #pragma unroll
    for (int m = 0; m < 4; m++) {
      int o = tid + m * 256; int h = o >> 6, dd = o & 63;
      float s = 0.0f;
      for (int t = 0; t < 32; t++) s += sc[h * 32 + t] * kbase[(size_t)t * 1024 + h * 64 + dd];
      z[o] = s + qh[o];
    }
    __syncthreads();
    // norm1 per head (d=64, ddof=1)
    if (tid < 16) {
      int h = tid; float mu = 0.0f, m2 = 0.0f;
      for (int d = 0; d < 64; d++) mu += z[h * 64 + d];
      mu *= (1.0f / 64.0f);
      for (int d = 0; d < 64; d++) { float v = z[h * 64 + d] - mu; m2 += v * v; }
      float inv = 1.0f / (sqrtf(m2 / 63.0f) + 1e-6f);
      for (int d = 0; d < 64; d++) {
        int idx = (i * 16 + h) * 64 + d;
        q_[h * 64 + d] = n1_a[idx] * (z[h * 64 + d] - mu) * inv + n1_b[idx];
      }
    }
    __syncthreads();
    // hid = relu(q_ @ ff_w1^T + b1)   [16][32]
    #pragma unroll
    for (int m = 0; m < 2; m++) {
      int o = tid + m * 256; int h = o >> 5, f = o & 31;
      const float* w = ff_w1 + ((size_t)(i * 16 + h) * 32 + f) * 64;
      float s = 0.0f;
      for (int d = 0; d < 64; d++) s += w[d] * q_[h * 64 + d];
      s += ff_b1[(i * 16 + h) * 32 + f];
      hid[o] = s > 0.0f ? s : 0.0f;
    }
    __syncthreads();
    // ffo + residual
    #pragma unroll
    for (int m = 0; m < 4; m++) {
      int o = tid + m * 256; int h = o >> 6, dd = o & 63;
      const float* w = ff_w2 + ((size_t)(i * 16 + h) * 64 + dd) * 32;
      float s = 0.0f;
      for (int f = 0; f < 32; f++) s += w[f] * hid[h * 32 + f];
      s += ff_b2[(i * 16 + h) * 64 + dd];
      z[o] = q_[o] + s;
    }
    __syncthreads();
    // norm2 per head -> q
    if (tid < 16) {
      int h = tid; float mu = 0.0f, m2 = 0.0f;
      for (int d = 0; d < 64; d++) mu += z[h * 64 + d];
      mu *= (1.0f / 64.0f);
      for (int d = 0; d < 64; d++) { float v = z[h * 64 + d] - mu; m2 += v * v; }
      float inv = 1.0f / (sqrtf(m2 / 63.0f) + 1e-6f);
      for (int d = 0; d < 64; d++) {
        int idx = (i * 16 + h) * 64 + d;
        q[h * 64 + d] = n2_a[idx] * (z[h * 64 + d] - mu) * inv + n2_b[idx];
      }
    }
    __syncthreads();
  }
  // final L2 normalize over 1024
  float ss = 0.0f;
  #pragma unroll
  for (int m = 0; m < 4; m++) { float v = q[tid + m * 256]; ss += v * v; }
  red[tid] = ss; __syncthreads();
  for (int st = 128; st > 0; st >>= 1) { if (tid < st) red[tid] += red[tid + st]; __syncthreads(); }
  float rn = 1.0f / fmaxf(sqrtf(red[0]), 1e-12f);
  #pragma unroll
  for (int m = 0; m < 4; m++) {
    int o = tid + m * 256;
    out[(size_t)b * 1024 + o] = q[o] * rn;
  }
}

// ---------------- launch ----------------
extern "C" void kernel_launch(void* const* d_in, const int* in_sizes, int n_in,
                              void* d_out, int out_size, void* d_ws, size_t ws_size,
                              hipStream_t stream) {
  (void)in_sizes; (void)n_in; (void)out_size; (void)ws_size;
  const float* x         = (const float*)d_in[0];
  const float* bn1_gamma = (const float*)d_in[1];
  const float* bn1_beta  = (const float*)d_in[2];
  const float* qpr_w     = (const float*)d_in[3];
  const float* qlin_w    = (const float*)d_in[4];
  const float* qlin_b    = (const float*)d_in[5];
  const float* n1_a      = (const float*)d_in[6];
  const float* n1_b      = (const float*)d_in[7];
  const float* n2_a      = (const float*)d_in[8];
  const float* n2_b      = (const float*)d_in[9];
  const float* ff_w1     = (const float*)d_in[10];
  const float* ff_b1     = (const float*)d_in[11];
  const float* ff_w2     = (const float*)d_in[12];
  const float* ff_b2     = (const float*)d_in[13];
  // d_in[14] (b=8), d_in[15] (t=32) are fixed scalars; hard-coded.

  char* ws = (char*)d_ws;
  bf16*  xs      = (bf16*) (ws);                            // 256*100352*2     = 51,380,224
  float* part    = (float*)(ws + 51380224);                 // 32*1024*256*4    = 33,554,432
  float* xq      = (float*)(ws + 51380224 + 33554432);      // 256*1024*4       =  1,048,576
  float* xqn     = (float*)(ws + 51380224 + 33554432 + 1048576);
  float* biastab = (float*)(ws + 51380224 + 33554432 + 2097152);  // 32*2048*4  =    262,144

  prep_bias   <<<256, 256, 0, stream>>>(bn1_beta, biastab);
  prep_x      <<<dim3(KTOT / 256, 256), 256, 0, stream>>>(x, bn1_gamma, biastab, xs);
  gemm_splitk <<<dim3(1024 / TILE_M, KSPLIT), 256, 0, stream>>>(qpr_w, xs, part);
  reduce_relu <<<1024, 256, 0, stream>>>(part, xq);
  layernorm_xq<<<256, 256, 0, stream>>>(xq, xqn);
  attn_tail   <<<8, 256, 0, stream>>>(xqn, qlin_w, qlin_b, n1_a, n1_b, n2_a, n2_b,
                                      ff_w1, ff_b1, ff_w2, ff_b2, (float*)d_out);
}
